// ODENetwork_14697378087662
// MI455X (gfx1250) — compile-verified, matrix operand order checked by reading, not run on hardware
//
#include <hip/hip_runtime.h>
#include <hip/hip_bf16.h>

typedef __attribute__((ext_vector_type(16))) _Float16 v16h;
typedef __attribute__((ext_vector_type(8)))  float    v8f;

__device__ __forceinline__ v8f wmma_f16(v16h a, v16h b, v8f c) {
  // D = A(16x32 f16) * B(32x16 f16) + C(16x16 f32)
  return __builtin_amdgcn_wmma_f32_16x16x32_f16(
      /*neg_a=*/false, a, /*neg_b=*/false, b,
      /*c_mod=*/(short)0, c, /*reuse_a=*/false, /*reuse_b=*/false);
}

// Branch-free tanh: prefer gfx1250 V_TANH; else raw v_exp/v_rcp construction.
__device__ __forceinline__ float fast_tanh(float x) {
#if __has_builtin(__builtin_amdgcn_tanhf)
  return __builtin_amdgcn_tanhf(x);
#else
  const float ax = __builtin_fabsf(x);
  // exp2(2*log2(e)*|x|) = e^(2|x|);  tanh(|x|) = 1 - 2/(e^(2|x|)+1)
  const float e = __builtin_amdgcn_exp2f(ax * 2.8853900817779268f);
  const float r = 1.0f - 2.0f * __builtin_amdgcn_rcpf(e + 1.0f);
  return __builtin_copysignf(r, x);
#endif
}

// Map K-index u (0..63) + row/point p to the half-index inside the per-channel
// A-fragment LDS block [kfrag(2)][lane(32)][16 halves] (ISA 16-bit A 16x32 layout).
__device__ __forceinline__ int a_idx(int u, int p) {
  int kfrag = u >> 5;
  int kk    = u & 31;
  int hi    = (kk >> 3) & 1;
  int j     = ((kk & 7) >> 1) + ((kk >> 4) << 2);
  int m     = kk & 1;
  int lane  = p + 16 * hi;
  return (kfrag * 32 + lane) * 16 + 2 * j + m;   // within one channel (1024 halves)
}

// tanh jet propagation. BND: {val, x, t, tt, xt, xtt}; else {val, dx, dv, dvv}.
template<int NC, bool BND>
__device__ __forceinline__ void tanh_jets(const float* __restrict__ z,
                                          float* __restrict__ hc) {
  const float h0 = fast_tanh(z[0]);
  const float s  = 1.0f - h0 * h0;
  hc[0] = h0;
  if (BND) {
    const float sp  = -2.0f * h0 * s;                 // d(sech^2)/da
    const float spp =  2.0f * s * (2.0f * h0 * h0 - s);
    hc[1] = s * z[1];
    hc[2] = s * z[2];
    hc[3] = s * z[3] + sp * z[2] * z[2];
    hc[4] = s * z[4] + sp * z[1] * z[2];
    hc[5] = s * z[5] + sp * (2.0f * z[4] * z[2] + z[1] * z[3])
                     + spp * z[1] * z[2] * z[2];
  } else {
    hc[1] = s * z[1];
    hc[2] = s * z[2];
    hc[3] = s * z[3] - 2.0f * h0 * s * z[2] * z[2];
  }
}

// One forward-jet pass through the 3-hidden-layer tanh MLP (16 points / wave).
template<int NC, bool BND>
__device__ void mlp_pass(int lane, float xin, float tin,
                         const _Float16* __restrict__ sB,
                         const float* __restrict__ sW1,
                         const float* __restrict__ sB1,
                         const float* __restrict__ sB2,
                         const float* __restrict__ sB3,
                         const float* __restrict__ sWo,
                         _Float16* __restrict__ sA,
                         float* __restrict__ sOut,
                         float* __restrict__ outacc)
{
  const int p = lane & 15;   // point within tile (pointwise partition)
  const int g = lane >> 4;   // unit half / accumulator row-half

  // ---- Layer 1: 2 -> 64, analytic input jets (K too small for WMMA) ----
#pragma clang loop unroll(disable)
  for (int uu = 0; uu < 32; ++uu) {
    const int u = g * 32 + uu;
    const float w0 = sW1[u], w1 = sW1[64 + u];
    float z[NC], hc[NC];
    z[0] = xin * w0 + tin * w1 + sB1[u];
    z[1] = w0;                              // d/dx of pre-activation
    z[2] = BND ? w1 : (w0 + w1);            // d/dt  or  d/dv (v=(1,1))
#pragma unroll
    for (int c = 3; c < NC; ++c) z[c] = 0.0f;
    tanh_jets<NC, BND>(z, hc);
    const int ai = a_idx(u, p);
#pragma unroll
    for (int c = 0; c < NC; ++c) sA[c * 1024 + ai] = (_Float16)hc[c];
  }
  __syncthreads();

  // ---- Hidden layers 2 & 3: WMMA, jets applied directly on accumulators ----
#pragma unroll
  for (int layer = 0; layer < 2; ++layer) {
    const _Float16* Bl = sB + layer * 4096;
    // Hoist all channels' A fragments (register-resident across the N loop).
    v16h a0[NC], a1[NC];
#pragma unroll
    for (int c = 0; c < NC; ++c) {
      a0[c] = *(const v16h*)(sA + c * 1024 +       lane * 16);
      a1[c] = *(const v16h*)(sA + c * 1024 + 512 + lane * 16);
    }
    __syncthreads();   // all A reads done before sA is overwritten below
    const float* bias = layer ? sB3 : sB2;
#pragma unroll
    for (int n = 0; n < 4; ++n) {
      const v16h b0 = *(const v16h*)(Bl + (0 * 4 + n) * 512 + lane * 16);
      const v16h b1 = *(const v16h*)(Bl + (1 * 4 + n) * 512 + lane * 16);
      v8f acc[NC];
#pragma unroll
      for (int c = 0; c < NC; ++c) {
        const v8f kz = {0.f, 0.f, 0.f, 0.f, 0.f, 0.f, 0.f, 0.f};
        acc[c] = wmma_f16(a0[c], b0, kz);
      }
#pragma unroll
      for (int c = 0; c < NC; ++c) acc[c] = wmma_f16(a1[c], b1, acc[c]);

      // D layout: element (row = v + 8*g, col = 16n + (lane&15)) in acc[.][v].
      // All channels share this mapping -> jet mixing is lane-local.
      const int col = n * 16 + (lane & 15);
      const float bz = bias[col];
#pragma unroll
      for (int v = 0; v < 8; ++v) {
        float z[NC], hc[NC];
#pragma unroll
        for (int c = 0; c < NC; ++c) z[c] = acc[c][v];
        z[0] += bz;
        tanh_jets<NC, BND>(z, hc);
        const int prow = v + 8 * g;
        if (layer == 0) {
          const int ai = a_idx(col, prow);   // col is next layer's K index
#pragma unroll
          for (int c = 0; c < NC; ++c) sA[c * 1024 + ai] = (_Float16)hc[c];
        } else {
          // final hidden activations staged for the Wo dot: [c][point][unit]
#pragma unroll
          for (int c = 0; c < NC; ++c) sOut[(c * 16 + prow) * 64 + col] = hc[c];
        }
      }
    }
    __syncthreads();
  }

  // ---- Output layer: u-jets = h3-jets . Wo ----
#pragma unroll
  for (int c = 0; c < NC; ++c) outacc[c] = 0.0f;
#pragma clang loop unroll(disable)
  for (int uu = 0; uu < 32; ++uu) {
    const int u = g * 32 + uu;
    const float wo = sWo[u];
#pragma unroll
    for (int c = 0; c < NC; ++c) outacc[c] += sOut[(c * 16 + p) * 64 + u] * wo;
  }
#pragma unroll
  for (int c = 0; c < NC; ++c) outacc[c] += __shfl_xor(outacc[c], 16, 32);
}

__global__ __launch_bounds__(32)
void pinn_residual_kernel(const float* __restrict__ points,
                          const float* __restrict__ W1, const float* __restrict__ b1,
                          const float* __restrict__ W2, const float* __restrict__ b2,
                          const float* __restrict__ W3, const float* __restrict__ b3,
                          const float* __restrict__ Wo, const float* __restrict__ bo,
                          float* __restrict__ partial, int npts)
{
  __shared__ __align__(32) _Float16 sB[2 * 4096];  // W2,W3 pre-swizzled B fragments
  __shared__ __align__(32) _Float16 sA[6 * 1024];  // activations as A fragments
  __shared__ float sOut[6 * 16 * 64];              // final-layer activations [c][p][u]
  __shared__ float sW1[128], sB1[64], sB2[64], sB3[64], sWo[64];
  __shared__ float sXT[2][16];

  const int lane = threadIdx.x;
  const int tile = blockIdx.x;

  if (lane == 0) {
    __builtin_prefetch(W2, 0, 3);
    __builtin_prefetch(W3, 0, 3);
    __builtin_prefetch(W1, 0, 3);
  }

#pragma clang loop unroll(disable)
  for (int i = lane; i < 128; i += 32) sW1[i] = W1[i];
#pragma clang loop unroll(disable)
  for (int i = lane; i < 64; i += 32) {
    sB1[i] = b1[i]; sB2[i] = b2[i]; sB3[i] = b3[i]; sWo[i] = Wo[i];
  }
  // Swizzle W2/W3 (row-major [K=in][N=out]) into per-lane B fragments.
  for (int l = 0; l < 2; ++l) {
    const float* Wsrc = l ? W3 : W2;
#pragma clang loop unroll(disable)
    for (int idx = lane; idx < 4096; idx += 32) {
      const int K = idx >> 6, col = idx & 63;
      const int kfrag = K >> 5, kk = K & 31;
      const int hi = (kk >> 3) & 1;
      const int j  = ((kk & 7) >> 1) + ((kk >> 4) << 2);
      const int m  = kk & 1;
      const int ntile = col >> 4, nn = col & 15;
      const int blane = nn + 16 * hi;
      sB[l * 4096 + (kfrag * 4 + ntile) * 512 + blane * 16 + 2 * j + m] =
          (_Float16)Wsrc[idx];
    }
  }
  if (lane < 16) {
    const int gp = tile * 16 + lane;
    const bool ok = gp < npts;
    sXT[0][lane] = ok ? points[gp * 2 + 0] : 0.0f;
    sXT[1][lane] = ok ? points[gp * 2 + 1] : 0.0f;
  }
  __syncthreads();

  const int p = lane & 15;
  const float xp = sXT[0][p], tp = sXT[1][p];

  // Boundary net at (1, t): jets {val, x, t, tt, xt, xtt}
  float obnd[6];
  mlp_pass<6, true>(lane, 1.0f, tp, sB, sW1, sB1, sB2, sB3, sWo, sA, sOut, obnd);
  // Interior net at (x, t): jets {val, d/dx, d/dv, d2/dv2}, v = (1,1)
  float oint[4];
  mlp_pass<4, false>(lane, xp, tp, sB, sW1, sB1, sB2, sB3, sWo, sA, sOut, oint);

  const float bov = bo[0];
  const float u    = oint[0] + bov, ux = oint[1], uv = oint[2], uvv = oint[3];
  const float u1   = obnd[0] + bov, u1x = obnd[1], u1t = obnd[2];
  const float u1tt = obnd[3], u1xt = obnd[4], u1xtt = obnd[5];

  // w(t) = u(1,t) + u_x(1,t) and its t-derivatives
  const float w   = u1 + u1x;
  const float wt  = u1t + u1xt;
  const float wtt = u1tt + u1xtt;

  const float PI = 3.14159265358979323846f;
  // v_sin/v_cos take revolutions (1.0 = 2*pi): sin(pi*t) = v_sin(t/2).
#if __has_builtin(__builtin_amdgcn_sinf) && __has_builtin(__builtin_amdgcn_cosf)
  const float snt = __builtin_amdgcn_sinf(0.5f * tp);
  const float cst = __builtin_amdgcn_cosf(0.5f * tp);
#else
  const float snt = sinf(PI * tp);
  const float cst = cosf(PI * tp);
#endif
  const float T  = tp * tp - tp,  Tp = 2.0f * tp - 1.0f;
  const float P  = xp * xp - xp,  Pp = 2.0f * xp - 1.0f;
  const float A  = T * P;
  const float B0 = u - w, Bx = ux, Bv = uv - wt, Bvv = uvv - wtt;

  const float psi  = 2.0f * xp * snt + A * B0;
  const float psix = 2.0f * snt + T * Pp * B0 + A * Bx;             // g1[0]
  const float Av   = Tp * P + T * Pp;
  const float Avv  = 2.0f * P + 2.0f * Tp * Pp + 2.0f * T;
  const float fvv  = 4.0f * PI * cst - 2.0f * PI * PI * xp * snt;   // (2x sin(pi t))_vv
  const float psivv = fvv + Avv * B0 + 2.0f * Av * Bv + A * Bvv;    // h[0]+h[1]

  const float forcing = snt * (2.0f - PI * PI * xp * xp + 2.0f * xp * xp * xp * snt);
  const float res = psivv + psi * psix - forcing;

  const bool valid = (lane < 16) && (tile * 16 + p < npts);
  float val = valid ? fabsf(res) : 0.0f;
#pragma unroll
  for (int off = 16; off >= 1; off >>= 1) val += __shfl_xor(val, off, 32);
  if (lane == 0) partial[tile] = val;
}

__global__ __launch_bounds__(256)
void reduce_kernel(const float* __restrict__ partial, float* __restrict__ out,
                   int ntiles, float scale)
{
  __shared__ float buf[256];
  float s = 0.0f;
#pragma clang loop unroll(disable)
  for (int i = threadIdx.x; i < ntiles; i += 256) s += partial[i]; // fixed order
  buf[threadIdx.x] = s;
  __syncthreads();
  for (int off = 128; off >= 1; off >>= 1) {
    if (threadIdx.x < off) buf[threadIdx.x] += buf[threadIdx.x + off];
    __syncthreads();
  }
  if (threadIdx.x == 0) out[0] = buf[0] * scale;
}

extern "C" void kernel_launch(void* const* d_in, const int* in_sizes, int n_in,
                              void* d_out, int out_size, void* d_ws, size_t ws_size,
                              hipStream_t stream) {
  const float* points = (const float*)d_in[0];
  const float* W1 = (const float*)d_in[1];
  const float* b1 = (const float*)d_in[2];
  const float* W2 = (const float*)d_in[3];
  const float* b2 = (const float*)d_in[4];
  const float* W3 = (const float*)d_in[5];
  const float* b3 = (const float*)d_in[6];
  const float* Wo = (const float*)d_in[7];
  const float* bo = (const float*)d_in[8];

  const int npts   = in_sizes[0] / 2;          // 262144
  const int ntiles = (npts + 15) / 16;         // 16384
  float* partial = (float*)d_ws;
  float* out = (float*)d_out;

  pinn_residual_kernel<<<ntiles, 32, 0, stream>>>(
      points, W1, b1, W2, b2, W3, b3, Wo, bo, partial, npts);
  reduce_kernel<<<1, 256, 0, stream>>>(partial, out, ntiles, 1.0f / (float)npts);
}
